// StandardGCN_7851200217634
// MI455X (gfx1250) — compile-verified, matrix operand order checked by reading, not run on hardware
//
#include <hip/hip_runtime.h>

// ---------------- problem constants (from reference) ----------------
#define NODES   100     // nodes per graph
#define PADN    112     // padded to 7 WMMA M-tiles of 16
#define HID     64
#define EPG     2000    // edges per graph
#define NGRAPH  1024
#define STRIDE  132     // feature row stride in LDS floats (128+4 -> bank spread)
#define LN_EPS  1e-5f

typedef _Float16 v16h __attribute__((ext_vector_type(16)));
typedef float    v8f  __attribute__((ext_vector_type(8)));

// ---------------- LDS layout (dynamic shared, 32B-aligned sections) ----
// bufX : PADN*STRIDE f32  (features / matmul out)   59136 B   @ 0
// bufY : PADN*STRIDE f32  (aggregation target)      59136 B   @ 59136
// wbuf : 128*128 f16      (W in fragment order)     32768 B   @ 118272
// abuf : 7*4*32*16 f16    (A in fragment order)     28672 B   @ 151040
// ew   : EPG f32                                     8000 B   @ 179712
// esrc : EPG u16                                     4000 B   @ 187712
// edst : EPG u16                                     4000 B   @ 191712
// snorm: PADN f32                                     448 B   @ 195712
// dnorm: PADN f32                                     448 B   @ 196160
#define SMEM_BYTES 196608   // = 192 KB  (fits 320 KB/WGP once)

__device__ __forceinline__ float wave_sum(float x) {
#pragma unroll
    for (int off = 16; off > 0; off >>= 1) x += __shfl_xor(x, off, 32);
    return x;
}

// One GraphConv layer, fully in LDS:
//   agg  = scatter_add( bufX[src] * src_norm[src] * ew )     (ds_add_f32 atomics)
//   bufX = LN+ReLU( (dst_norm * agg) @ W + b )               (v_wmma_f32_16x16x32_f16,
//                                                             operands pre-staged in
//                                                             fragment order -> b128 LDS loads)
template <int DIN, int DOUT>
__device__ void gcn_layer(float* __restrict__ bufX, float* __restrict__ bufY,
                          _Float16* __restrict__ wbuf, _Float16* __restrict__ abuf,
                          const float* __restrict__ snorm, const float* __restrict__ dnorm,
                          const unsigned short* __restrict__ esrc,
                          const unsigned short* __restrict__ edst,
                          const float* __restrict__ ew,
                          const float* __restrict__ W, const float* __restrict__ bias,
                          const float* __restrict__ gamma, const float* __restrict__ beta,
                          int tid)
{
    const int bd = blockDim.x;
    constexpr int NT = DOUT >> 4;           // N tiles
    constexpr int KS = DIN >> 5;            // K=32 steps

    // ---- A) zero agg buffer + stage W as f16 in WMMA B-fragment order ----
    // B fragment layout: wbuf[((nt*KS+ks)*32 + lane)*16 + i], lane = hi*16 + (n&15),
    //   lanes 0-15 hold K = {kb..kb+7, kb+16..kb+23}, lanes 16-31 shifted by 8.
    for (int i = tid; i < PADN * STRIDE; i += bd) bufY[i] = 0.f;
    for (int idx = tid; idx < DIN * DOUT; idx += bd) {
        int k  = idx / DOUT;
        int n  = idx - k * DOUT;
        int ks = k >> 5;
        int kk = k & 31;
        int hi = (kk >> 3) & 1;
        int i  = (kk & 7) + ((kk >> 4) << 3);
        int lane = hi * 16 + (n & 15);
        int nt = n >> 4;
        wbuf[(((nt * KS + ks) * 32) + lane) * 16 + i] = (_Float16)W[idx];
    }
    __syncthreads();

    // ---- B) edge aggregation: gather bufX, scatter-add into bufY ----
    constexpr int CH = DIN >> 2;            // float4 chunks per edge
    for (int idx = tid; idx < EPG * CH; idx += bd) {
        int e  = idx / CH;
        int k0 = (idx - e * CH) << 2;
        int s  = esrc[e];
        int d  = edst[e];
        float w = ew[e] * snorm[s];
        const float* hs = bufX + s * STRIDE + k0;
        float*       ad = bufY + d * STRIDE + k0;
        atomicAdd(ad + 0, hs[0] * w);       // ds_add_f32 (no return)
        atomicAdd(ad + 1, hs[1] * w);
        atomicAdd(ad + 2, hs[2] * w);
        atomicAdd(ad + 3, hs[3] * w);
    }
    __syncthreads();

    // ---- B2) stage A = (dst_norm * agg) as f16 in WMMA A-fragment order ----
    // abuf[((mt*KS+ks)*32 + lane)*16 + i]  <-  bufY[row][k] * dnorm[row]
    for (int idx = tid; idx < 7 * KS * 512; idx += bd) {
        int f    = idx >> 9;                // fragment = mt*KS + ks
        int r    = idx & 511;
        int lane = r >> 4;
        int i    = r & 15;
        int mt   = f / KS;
        int ks   = f - mt * KS;
        int hi   = lane >> 4;
        int row  = mt * 16 + (lane & 15);
        int kb   = ks * 32 + hi * 8;
        int k    = (i < 8) ? (kb + i) : (kb + 8 + i);   // kb+16+(i-8)
        abuf[idx] = (_Float16)(bufY[row * STRIDE + k] * dnorm[row]);
    }
    __syncthreads();

    // ---- C) WMMA matmul: bufX[112,DOUT] = A @ W + b ----
    const int lane   = tid & 31;
    const int wid    = tid >> 5;
    const int nwaves = bd >> 5;
    const int r16 = lane & 15;
    const int hi  = lane >> 4;
    const _Float16* apl = abuf + lane * 16; // per-lane base, frag f at +f*512
    const _Float16* bpl = wbuf + lane * 16;

    for (int t = wid; t < 7 * NT; t += nwaves) {
        int mt = t / NT, nt = t - mt * NT;
        v8f acc = {};
#pragma unroll
        for (int ks = 0; ks < KS; ++ks) {
            v16h a = *(const v16h*)(apl + (mt * KS + ks) * 512);
            v16h b = *(const v16h*)(bpl + (nt * KS + ks) * 512);
            acc = __builtin_amdgcn_wmma_f32_16x16x32_f16(
                false, a, false, b, (short)0, acc, false, false);
        }
        // D layout: VGPR r -> row (mt*16 + hi*8 + r), col (nt*16 + lane&15)
        int ocol  = nt * 16 + r16;
        int rbase = mt * 16 + hi * 8;
        float bv = bias[ocol];
#pragma unroll
        for (int r = 0; r < 8; ++r)
            bufX[(rbase + r) * STRIDE + ocol] = acc[r] + bv;
    }
    __syncthreads();

    // ---- D) LayerNorm + ReLU on rows 0..99, zero pad rows ----
    constexpr int NV = DOUT >> 5;           // values per lane (2 or 4)
    for (int row = wid; row < PADN; row += nwaves) {
        float* rp = bufX + row * STRIDE;
        if (row < NODES) {
            float vals[NV];
            float s = 0.f, sq = 0.f;
#pragma unroll
            for (int j = 0; j < NV; ++j) {
                float v = rp[lane + j * 32];
                vals[j] = v; s += v; sq += v * v;
            }
            s  = wave_sum(s);
            sq = wave_sum(sq);
            float mu   = s * (1.f / DOUT);
            float var  = sq * (1.f / DOUT) - mu * mu;
            float rstd = rsqrtf(var + LN_EPS);
#pragma unroll
            for (int j = 0; j < NV; ++j) {
                int k = lane + j * 32;
                float y = (vals[j] - mu) * rstd * gamma[k] + beta[k];
                rp[k] = y > 0.f ? y : 0.f;
            }
        } else {
#pragma unroll
            for (int j = 0; j < NV; ++j) rp[lane + j * 32] = 0.f;
        }
    }
    __syncthreads();
}

__global__ void __launch_bounds__(256)
gcn_fused_kernel(const int* __restrict__ src, const int* __restrict__ dst,
                 const float* __restrict__ ewg, const float* __restrict__ nemb,
                 const float* __restrict__ W1, const float* __restrict__ b1,
                 const float* __restrict__ g1, const float* __restrict__ be1,
                 const float* __restrict__ W2, const float* __restrict__ b2,
                 const float* __restrict__ g2, const float* __restrict__ be2,
                 const float* __restrict__ W3, const float* __restrict__ b3,
                 const float* __restrict__ g3, const float* __restrict__ be3,
                 const float* __restrict__ Wfc, const float* __restrict__ bfc,
                 float* __restrict__ out)
{
    extern __shared__ char smem[];
    float*          bufX  = (float*)smem;                       // 59136 B
    float*          bufY  = bufX + PADN * STRIDE;               // 59136 B
    _Float16*       wbuf  = (_Float16*)(bufY + PADN * STRIDE);  // 32768 B
    _Float16*       abuf  = wbuf + 128 * 128;                   // 28672 B
    float*          ew    = (float*)(abuf + 7 * 4 * 32 * 16);   //  8000 B
    unsigned short* esrc  = (unsigned short*)(ew + EPG);        //  4000 B
    unsigned short* edst  = esrc + EPG;                         //  4000 B
    float*          snorm = (float*)(edst + EPG);               //   448 B
    float*          dnorm = snorm + PADN;                       //   448 B

    const int g   = blockIdx.x;
    const int tid = threadIdx.x;
    const int bd  = blockDim.x;
    const int base = g * EPG;
    const int goff = g * NODES;

    // ---- setup: degrees + edge staging + node embedding ----
    for (int i = tid; i < PADN; i += bd) { snorm[i] = 0.f; dnorm[i] = 0.f; }
    for (int i = tid; i < PADN * HID; i += bd) {
        int row = i / HID, k = i - row * HID;
        bufX[row * STRIDE + k] = (row < NODES) ? nemb[row * HID + k] : 0.f;
    }
    __syncthreads();

    for (int e = tid; e < EPG; e += bd) {
        // prefetch next strip of the only HBM-streamed per-graph data
        if (e + bd < EPG) {
            __builtin_prefetch(&src[base + e + bd], 0, 0);  // global_prefetch_b8
            __builtin_prefetch(&dst[base + e + bd], 0, 0);
            __builtin_prefetch(&ewg[base + e + bd], 0, 0);
        }
        int s = src[base + e] - goff;
        int d = dst[base + e] - goff;
        esrc[e] = (unsigned short)s;
        edst[e] = (unsigned short)d;
        ew[e]   = ewg[base + e];
        atomicAdd(&snorm[s], 1.f);          // out-degree
        atomicAdd(&dnorm[d], 1.f);          // in-degree
    }
    __syncthreads();
    for (int i = tid; i < PADN; i += bd) {
        snorm[i] = (i < NODES) ? rsqrtf(fmaxf(snorm[i], 1.f)) : 0.f;
        dnorm[i] = (i < NODES) ? rsqrtf(fmaxf(dnorm[i], 1.f)) : 0.f;
    }
    __syncthreads();

    // ---- 3 fused GraphConv + LN + ReLU layers ----
    gcn_layer< 64, 128>(bufX, bufY, wbuf, abuf, snorm, dnorm, esrc, edst, ew, W1, b1, g1, be1, tid);
    gcn_layer<128, 128>(bufX, bufY, wbuf, abuf, snorm, dnorm, esrc, edst, ew, W2, b2, g2, be2, tid);
    gcn_layer<128,  64>(bufX, bufY, wbuf, abuf, snorm, dnorm, esrc, edst, ew, W3, b3, g3, be3, tid);

    // ---- mean pool over 100 nodes + FC(64 -> 2), wave 0 only ----
    if (tid < 32) {
        int lane = tid;
        float s0 = 0.f, s1 = 0.f;
        for (int row = 0; row < NODES; ++row) {
            s0 += bufX[row * STRIDE + lane];
            s1 += bufX[row * STRIDE + lane + 32];
        }
        s0 *= (1.f / NODES);
        s1 *= (1.f / NODES);
        float p0 = s0 * Wfc[lane * 2 + 0] + s1 * Wfc[(lane + 32) * 2 + 0];
        float p1 = s0 * Wfc[lane * 2 + 1] + s1 * Wfc[(lane + 32) * 2 + 1];
        p0 = wave_sum(p0);
        p1 = wave_sum(p1);
        if (lane == 0) {
            out[g * 2 + 0] = p0 + bfc[0];
            out[g * 2 + 1] = p1 + bfc[1];
        }
    }
}

extern "C" void kernel_launch(void* const* d_in, const int* in_sizes, int n_in,
                              void* d_out, int out_size, void* d_ws, size_t ws_size,
                              hipStream_t stream) {
    (void)in_sizes; (void)n_in; (void)out_size; (void)d_ws; (void)ws_size;
    const int*   src  = (const int*)d_in[0];
    const int*   dst  = (const int*)d_in[1];
    const float* ew   = (const float*)d_in[2];
    // d_in[3] = batch_size scalar (fixed 1024 in reference)
    const float* nemb = (const float*)d_in[4];
    const float* W1   = (const float*)d_in[5];
    const float* b1   = (const float*)d_in[6];
    const float* g1   = (const float*)d_in[7];
    const float* be1  = (const float*)d_in[8];
    const float* W2   = (const float*)d_in[9];
    const float* b2   = (const float*)d_in[10];
    const float* g2   = (const float*)d_in[11];
    const float* be2  = (const float*)d_in[12];
    const float* W3   = (const float*)d_in[13];
    const float* b3   = (const float*)d_in[14];
    const float* g3   = (const float*)d_in[15];
    const float* be3  = (const float*)d_in[16];
    const float* Wfc  = (const float*)d_in[17];
    const float* bfc  = (const float*)d_in[18];
    float* out = (float*)d_out;

    hipLaunchKernelGGL(gcn_fused_kernel, dim3(NGRAPH), dim3(256), SMEM_BYTES, stream,
                       src, dst, ew, nemb,
                       W1, b1, g1, be1, W2, b2, g2, be2, W3, b3, g3, be3,
                       Wfc, bfc, out);
}
